// CrossViewAttention_36893769073121
// MI455X (gfx1250) — matrix operand order repaired and checked by reading, not compiled, not run on hardware
//
#include <hip/hip_runtime.h>
#include <hip/hip_bf16.h>
#include <math.h>

typedef __attribute__((ext_vector_type(16))) _Float16 v16h;
typedef __attribute__((ext_vector_type(8)))  float    v8f;
typedef __attribute__((ext_vector_type(4)))  unsigned int u32x4;
typedef __attribute__((ext_vector_type(8)))  unsigned int u32x8;

#define NBATCH 4
#define NVIEW  6
#define CCH    128
#define DDM    128
#define QQ     625
#define KPP    1152          // fh*fw
#define NKP    6912          // NVIEW*KPP
#define BNIMG  24            // B*N
#define MKV    27648         // BNIMG*KPP rows for k/v
#define MQ     2500          // B*Q rows for q
#define EAF_ZMAX 4.0f

// ---------------------------------------------------------------------------
// elementwise helpers
// ---------------------------------------------------------------------------
__global__ void f32_to_f16_kernel(const float* __restrict__ src, _Float16* __restrict__ dst, int n) {
    int i = blockIdx.x * blockDim.x + threadIdx.x;
    if (i < n) dst[i] = (_Float16)src[i];
}

// feature (BN, C, P) f32 -> BN+ReLU -> featH (BN, P, C) f16
__global__ void bn_relu_tr_kernel(const float* __restrict__ f,
                                  const float* __restrict__ g, const float* __restrict__ b,
                                  const float* __restrict__ m, const float* __restrict__ var,
                                  _Float16* __restrict__ out) {
    int idx = blockIdx.x * blockDim.x + threadIdx.x;
    int total = BNIMG * CCH * KPP;
    if (idx >= total) return;
    int p  = idx % KPP;
    int c  = (idx / KPP) % CCH;
    int bn = idx / (KPP * CCH);
    float y = (f[idx] - m[c]) * rsqrtf(var[c] + 1e-5f) * g[c] + b[c];
    y = fmaxf(y, 0.f);
    out[((size_t)(bn * KPP + p)) * CCH + c] = (_Float16)y;
}

// ---------------------------------------------------------------------------
// generic WMMA GEMM:  Y[M,N] = act( X[M,K] * W[N,K]^T + bias[N] )
// ---------------------------------------------------------------------------
__global__ void gemm_xwT_kernel(const _Float16* __restrict__ X, const _Float16* __restrict__ W,
                                const float* __restrict__ bias, void* __restrict__ Y,
                                int M, int Nn, int K, int act, int out_half) {
    int lane = threadIdx.x & 31;
    int wave = threadIdx.x >> 5;
    int ntn  = Nn >> 4;
    int mtiles = (M + 15) >> 4;
    int tile = blockIdx.x * (blockDim.x >> 5) + wave;
    int mt = tile / ntn, nt = tile % ntn;
    if (mt >= mtiles) return;

    int lo = lane & 15;
    bool lower = lane < 16;
    int m0 = mt * 16, n0 = nt * 16;
    int row = m0 + lo;
    bool rvalid = row < M;
    int col = n0 + lo;

    int kbA = lower ? 0 : 16;
    int kbB = lower ? 0 : 16;
    const _Float16* xp = X + (size_t)row * K;
    const _Float16* wp = W + (size_t)col * K;

    v8f acc = {};
    for (int k0 = 0; k0 < K; k0 += 32) {
        __builtin_prefetch(wp + k0 + 64, 0, 0);   // global_prefetch_b8
        v16h a = {};
        if (rvalid) {
#pragma unroll
            for (int i = 0; i < 8; ++i) a[i]     = xp[k0 + kbA + i];
#pragma unroll
            for (int i = 0; i < 8; ++i) a[8 + i] = xp[k0 + kbA + 8 + i];
        }
        v16h bb;
#pragma unroll
        for (int i = 0; i < 16; ++i) bb[i] = wp[k0 + kbB + i];
        acc = __builtin_amdgcn_wmma_f32_16x16x32_f16(false, a, false, bb, (short)0, acc, false, false);
    }

    float bv = bias ? bias[col] : 0.f;
    int hi8 = lower ? 0 : 8;
#pragma unroll
    for (int i = 0; i < 8; ++i) {
        int r = m0 + i + hi8;
        if (r >= M) continue;
        float y = acc[i] + bv;
        if (act == 1)      y = fmaxf(y, 0.f);
        else if (act == 2) y = 0.5f * y * (1.f + erff(y * 0.70710678f));   // exact GELU
        if (out_half) ((_Float16*)Y)[(size_t)r * Nn + col] = (_Float16)y;
        else          ((float*)Y)[(size_t)r * Nn + col] = y;
    }
}

// ---------------------------------------------------------------------------
// row LayerNorm (cols = 128), one wave per row
// ---------------------------------------------------------------------------
__global__ void ln128_kernel(const float* __restrict__ X, const float* __restrict__ g,
                             const float* __restrict__ b, _Float16* __restrict__ Yh, int rows) {
    int wv = (blockIdx.x * blockDim.x + threadIdx.x) >> 5;
    int lane = threadIdx.x & 31;
    if (wv >= rows) return;
    const float* xp = X + (size_t)wv * 128;
    float v[4], s = 0.f;
#pragma unroll
    for (int i = 0; i < 4; ++i) { v[i] = xp[i * 32 + lane]; s += v[i]; }
    for (int o = 16; o > 0; o >>= 1) s += __shfl_xor(s, o, 32);
    float mean = s * (1.f / 128.f), vr = 0.f;
#pragma unroll
    for (int i = 0; i < 4; ++i) { float d = v[i] - mean; vr += d * d; }
    for (int o = 16; o > 0; o >>= 1) vr += __shfl_xor(vr, o, 32);
    float inv = rsqrtf(vr * (1.f / 128.f) + 1e-5f);
#pragma unroll
    for (int i = 0; i < 4; ++i) {
        int c = i * 32 + lane;
        Yh[(size_t)wv * 128 + c] = (_Float16)((v[i] - mean) * inv * g[c] + b[c]);
    }
}

// q rows gathered from x (B, D, Q) -> LN -> f16
__global__ void ln_q_kernel(const float* __restrict__ x, const float* __restrict__ g,
                            const float* __restrict__ b, _Float16* __restrict__ Yh) {
    int wv = (blockIdx.x * blockDim.x + threadIdx.x) >> 5;
    int lane = threadIdx.x & 31;
    if (wv >= MQ) return;
    int bb = wv / QQ, q = wv % QQ;
    float v[4], s = 0.f;
#pragma unroll
    for (int i = 0; i < 4; ++i) {
        int c = i * 32 + lane;
        v[i] = x[((size_t)(bb * DDM + c)) * QQ + q];
        s += v[i];
    }
    for (int o = 16; o > 0; o >>= 1) s += __shfl_xor(s, o, 32);
    float mean = s * (1.f / 128.f), vr = 0.f;
#pragma unroll
    for (int i = 0; i < 4; ++i) { float d = v[i] - mean; vr += d * d; }
    for (int o = 16; o > 0; o >>= 1) vr += __shfl_xor(vr, o, 32);
    float inv = rsqrtf(vr * (1.f / 128.f) + 1e-5f);
#pragma unroll
    for (int i = 0; i < 4; ++i) {
        int c = i * 32 + lane;
        Yh[(size_t)wv * 128 + c] = (_Float16)((v[i] - mean) * inv * g[c] + b[c]);
    }
}

// z = o_raw + qt(x gather); LN(ln_pre); store f32 + f16
__global__ void resid_ln_kernel(const float* __restrict__ o_raw, const float* __restrict__ x,
                                const float* __restrict__ g, const float* __restrict__ b,
                                float* __restrict__ zf, _Float16* __restrict__ zh) {
    int wv = (blockIdx.x * blockDim.x + threadIdx.x) >> 5;
    int lane = threadIdx.x & 31;
    if (wv >= MQ) return;
    int bb = wv / QQ, q = wv % QQ;
    float v[4], s = 0.f;
#pragma unroll
    for (int i = 0; i < 4; ++i) {
        int c = i * 32 + lane;
        v[i] = o_raw[(size_t)wv * 128 + c] + x[((size_t)(bb * DDM + c)) * QQ + q];
        s += v[i];
    }
    for (int o = 16; o > 0; o >>= 1) s += __shfl_xor(s, o, 32);
    float mean = s * (1.f / 128.f), vr = 0.f;
#pragma unroll
    for (int i = 0; i < 4; ++i) { float d = v[i] - mean; vr += d * d; }
    for (int o = 16; o > 0; o >>= 1) vr += __shfl_xor(vr, o, 32);
    float inv = rsqrtf(vr * (1.f / 128.f) + 1e-5f);
#pragma unroll
    for (int i = 0; i < 4; ++i) {
        int c = i * 32 + lane;
        float y = (v[i] - mean) * inv * g[c] + b[c];
        zf[(size_t)wv * 128 + c] = y;
        zh[(size_t)wv * 128 + c] = (_Float16)y;
    }
}

// z2 = z_ln + m2; LN(ln_post); scatter to out (B, D, Q)
__global__ void final_ln_kernel(const float* __restrict__ z_ln, const float* __restrict__ m2,
                                const float* __restrict__ g, const float* __restrict__ b,
                                float* __restrict__ out) {
    int wv = (blockIdx.x * blockDim.x + threadIdx.x) >> 5;
    int lane = threadIdx.x & 31;
    if (wv >= MQ) return;
    int bb = wv / QQ, q = wv % QQ;
    float v[4], s = 0.f;
#pragma unroll
    for (int i = 0; i < 4; ++i) {
        int c = i * 32 + lane;
        v[i] = z_ln[(size_t)wv * 128 + c] + m2[(size_t)wv * 128 + c];
        s += v[i];
    }
    for (int o = 16; o > 0; o >>= 1) s += __shfl_xor(s, o, 32);
    float mean = s * (1.f / 128.f), vr = 0.f;
#pragma unroll
    for (int i = 0; i < 4; ++i) { float d = v[i] - mean; vr += d * d; }
    for (int o = 16; o > 0; o >>= 1) vr += __shfl_xor(vr, o, 32);
    float inv = rsqrtf(vr * (1.f / 128.f) + 1e-5f);
#pragma unroll
    for (int i = 0; i < 4; ++i) {
        int c = i * 32 + lane;
        out[((size_t)(bb * DDM + c)) * QQ + q] = (v[i] - mean) * inv * g[c] + b[c];
    }
}

// ---------------------------------------------------------------------------
// geometry prep
// ---------------------------------------------------------------------------
__global__ void prep_mats_kernel(const float* __restrict__ E_inv, const float* __restrict__ I_inv,
                                 float* __restrict__ Eb, float* __restrict__ Ib) {
    int idx = threadIdx.x;
    if (idx >= BNIMG) return;
    const float* Ei = E_inv + idx * 16;
    float R[3][3], t3[3];
#pragma unroll
    for (int r = 0; r < 3; ++r) {
#pragma unroll
        for (int c = 0; c < 3; ++c) R[r][c] = Ei[r * 4 + c];
        t3[r] = Ei[r * 4 + 3];
    }
#pragma unroll
    for (int r = 0; r < 3; ++r) {
        Eb[idx * 12 + r * 4 + 0] = R[0][r];
        Eb[idx * 12 + r * 4 + 1] = R[1][r];
        Eb[idx * 12 + r * 4 + 2] = R[2][r];
        Eb[idx * 12 + r * 4 + 3] = -(R[0][r] * t3[0] + R[1][r] * t3[1] + R[2][r] * t3[2]);
    }
    const float* Ii = I_inv + idx * 9;
    float a = Ii[0], b = Ii[1], c = Ii[2], d = Ii[3], e = Ii[4], f = Ii[5],
          g = Ii[6], h = Ii[7], i9 = Ii[8];
    float det = a * (e * i9 - f * h) - b * (d * i9 - f * g) + c * (d * h - e * g);
    float id = 1.f / det;
    float* I = Ib + idx * 9;
    I[0] = (e * i9 - f * h) * id; I[1] = (c * h - b * i9) * id; I[2] = (b * f - c * e) * id;
    I[3] = (f * g - d * i9) * id; I[4] = (a * i9 - c * g) * id; I[5] = (c * d - a * f) * id;
    I[6] = (d * h - e * g) * id;  I[7] = (b * g - a * h) * id;  I[8] = (a * e - b * d) * id;
}

__global__ void distmax_kernel(const float* __restrict__ E_inv, const float* __restrict__ bev,
                               float* __restrict__ dmax) {
    int b = blockIdx.x, tid = threadIdx.x;
    float m = 0.f;
    for (int idx = tid; idx < NVIEW * QQ; idx += 256) {
        int n = idx / QQ, q = idx % QQ;
        const float* Ei = E_inv + (size_t)(b * NVIEW + n) * 16;
        float dx = bev[q] - Ei[3], dy = bev[QQ + q] - Ei[7];
        m = fmaxf(m, sqrtf(dx * dx + dy * dy) + 1e-6f);
    }
    __shared__ float red[256];
    red[tid] = m;
    __syncthreads();
    for (int s = 128; s > 0; s >>= 1) {
        if (tid < s) red[tid] = fmaxf(red[tid], red[tid + s]);
        __syncthreads();
    }
    if (tid == 0) dmax[b] = red[0];
}

__global__ void geom_kernel(const float* __restrict__ bev, const float* __restrict__ Eb,
                            const float* __restrict__ Ib, const float* __restrict__ E_inv,
                            const float* __restrict__ dmax, float4* __restrict__ G) {
    int tid = blockIdx.x * blockDim.x + threadIdx.x;
    if (tid >= BNIMG * QQ) return;
    int q = tid % QQ, bn = tid / QQ, b = bn / NVIEW;
    float gx = bev[q], gy = bev[QQ + q];
    const float* e = Eb + bn * 12;
    const float* Ic = Ib + bn * 9;
    float P0[3], P1[3];
#pragma unroll
    for (int r = 0; r < 3; ++r) {
        float base = e[r * 4 + 0] * gx + e[r * 4 + 1] * gy + e[r * 4 + 3];
        P0[r] = base;
        P1[r] = base + e[r * 4 + 2] * EAF_ZMAX;
    }
    float p0[3], p1[3];
#pragma unroll
    for (int r = 0; r < 3; ++r) {
        p0[r] = Ic[r * 3 + 0] * P0[0] + Ic[r * 3 + 1] * P0[1] + Ic[r * 3 + 2] * P0[2];
        p1[r] = Ic[r * 3 + 0] * P1[0] + Ic[r * 3 + 1] * P1[1] + Ic[r * 3 + 2] * P1[2];
    }
    float z0 = p0[2] + 1e-8f, z1 = p1[2] + 1e-8f;
#pragma unroll
    for (int r = 0; r < 3; ++r) { p0[r] /= z0; p1[r] /= z1; }
    float lx = p0[1] * p1[2] - p0[2] * p1[1];
    float ly = p0[2] * p1[0] - p0[0] * p1[2];
    float lz = p0[0] * p1[1] - p0[1] * p1[0];
    float den = fmaxf(sqrtf(lx * lx + ly * ly), 1e-8f);
    const float* Ei = E_inv + (size_t)bn * 16;
    float dx = gx - Ei[3], dy = gy - Ei[7];
    float dist = sqrtf(dx * dx + dy * dy) + 1e-6f;
    float dn = fminf(fmaxf(dist / (dmax[b] + 1e-6f), 0.f), 1.f);
    float sigma = 8.f - dn * 7.f;
    float lam = 1.f / (sigma + 1e-6f);
    float4 r4;
    r4.x = lx / den; r4.y = ly / den; r4.z = lz / den; r4.w = lam * lam;
    G[tid] = r4;
}

__global__ void ximg_kernel(const float* __restrict__ ip, float4* __restrict__ xo) {
    int k = blockIdx.x * blockDim.x + threadIdx.x;
    if (k >= KPP) return;
    float4 r;
    r.x = ip[k]; r.y = ip[KPP + k]; r.z = ip[2 * KPP + k]; r.w = 0.f;
    xo[k] = r;
}

// ---------------------------------------------------------------------------
// CDNA5 data movers
// ---------------------------------------------------------------------------
// TDM: load one 2-D tile (128 halfs wide x 32 rows, row stride 128 halfs)
// from global to LDS. D# per CDNA5 ISA §8.3/§8.4 (2-group form).
__device__ __forceinline__ void tdm_load_tile_128x32(unsigned lds_off, const void* gsrc) {
    unsigned long long va = (unsigned long long)(uintptr_t)gsrc;
    u32x4 g0;
    g0[0] = 1u;                                   // count=1, user mode
    g0[1] = lds_off;                              // lds_addr (bytes)
    g0[2] = (unsigned)va;                         // global_addr[31:0]
    g0[3] = (unsigned)((va >> 32) & 0x01FFFFFFull) | (2u << 30);  // addr[56:32] | type=2
    u32x8 g1;
    g1[0] = 1u << 16;        // data_size=1 (2 bytes)
    g1[1] = 128u << 16;      // tensor_dim0 = 128 elems
    g1[2] = 32u << 16;       // tensor_dim1 = 32 rows
    g1[3] = 128u << 16;      // tile_dim0 = 128
    g1[4] = 32u;             // tile_dim1 = 32, tile_dim2 = 0
    g1[5] = 128u;            // tensor_dim0_stride = 128 elems
    g1[6] = 0u;
    g1[7] = 0u;
    asm volatile("tensor_load_to_lds %0, %1" :: "s"(g0), "s"(g1) : "memory");
}

// async global->LDS copy of an 8 KB tile, 16 B per lane per issue (4 issues)
__device__ __forceinline__ void async_copy_tile_8k(unsigned lds_base, const void* gsrc, int tid) {
    unsigned long long base = (unsigned long long)(uintptr_t)gsrc;
#pragma unroll
    for (int j = 0; j < 4; ++j) {
        unsigned off = (unsigned)(tid + j * 128) * 16u;
        unsigned lds = lds_base + off;
        asm volatile("global_load_async_to_lds_b128 %0, %1, %2"
                     :: "v"(lds), "v"(off), "s"(base) : "memory");
    }
}

// ---------------------------------------------------------------------------
// flash attention with epipolar mask
// one workgroup per (b, 16-row q tile); 4 waves = 4 heads share LDS K/V tiles
// K tile staged by the Tensor Data Mover, V tile by async global->LDS copies,
// double-buffered (TENSORcnt / ASYNCcnt pipelining).
// ---------------------------------------------------------------------------
__global__ __launch_bounds__(128) void attn_kernel(
        const _Float16* __restrict__ qpH, const _Float16* __restrict__ kpH,
        const _Float16* __restrict__ vpH, const float4* __restrict__ G,
        const float4* __restrict__ ximg, _Float16* __restrict__ attn_outH) {
    __shared__ __align__(64) _Float16 kTile[2][32][128];   // 2 x 8 KB
    __shared__ __align__(64) _Float16 vTile[2][32][128];   // 2 x 8 KB
    __shared__ _Float16 pbuf[4][16][40];

    const int CHUNKS = NKP / 32;   // 216
    const int CPV    = KPP / 32;   // 36 chunks per view

    int tid  = threadIdx.x;
    int lane = tid & 31;
    int h    = tid >> 5;           // wave id == head
    int b    = blockIdx.y;
    int q0   = blockIdx.x * 16;
    int lo   = lane & 15;
    bool lower = lane < 16;
    int hi8 = lower ? 0 : 8;
    int kbA = lower ? 0 : 16;
    int kbB = lower ? 0 : 16;

    auto stage = [&](int chunk, int buf) {
        const _Float16* ks = kpH + ((size_t)(b * NKP + chunk * 32)) * 128;
        const _Float16* vs = vpH + ((size_t)(b * NKP + chunk * 32)) * 128;
        if (h == 0) tdm_load_tile_128x32((unsigned)(uintptr_t)&kTile[buf][0][0], ks);
        async_copy_tile_8k((unsigned)(uintptr_t)&vTile[buf][0][0], vs, tid);
    };

    // resident q fragment (rows q0..q0+15, dims h*32..h*32+31)
    v16h aq = {};
    {
        int qr = q0 + lo;
        if (qr < QQ) {
            const _Float16* qp = qpH + ((size_t)(b * QQ + qr)) * 128 + h * 32;
#pragma unroll
            for (int i = 0; i < 8; ++i) aq[i]     = qp[kbA + i];
#pragma unroll
            for (int i = 0; i < 8; ++i) aq[8 + i] = qp[kbA + 8 + i];
        }
    }

    float Mrun[8], Srun[8];
    v8f O0 = {}, O1 = {};
#pragma unroll
    for (int i = 0; i < 8; ++i) { Mrun[i] = -1e30f; Srun[i] = 0.f; }
    const float invs = 0.1767766953f;   // 1/sqrt(32)

    stage(0, 0);

    int ncur = -1;
    float4 gf[8];
    for (int c = 0; c < CHUNKS; ++c) {
        int buf = c & 1;
        if (c + 1 < CHUNKS) {
            stage(c + 1, buf ^ 1);
            asm volatile("s_wait_asynccnt 0x4" ::: "memory");   // prev 4 async done
            if (h == 0) __builtin_amdgcn_s_wait_tensorcnt(1);   // prev TDM done
        } else {
            asm volatile("s_wait_asynccnt 0x0" ::: "memory");
            if (h == 0) __builtin_amdgcn_s_wait_tensorcnt(0);
        }
        __syncthreads();

        int n = c / CPV;
        if (n != ncur) {
            ncur = n;
#pragma unroll
            for (int i = 0; i < 8; ++i) {
                int qr = q0 + i + hi8;
                if (qr >= QQ) qr = QQ - 1;
                gf[i] = G[(size_t)((b * NVIEW + n) * QQ + qr)];
            }
        }

        float raw[2][8];
#pragma unroll
        for (int sub = 0; sub < 2; ++sub) {
            v16h bb;
#pragma unroll
            for (int i = 0; i < 16; ++i) bb[i] = kTile[buf][sub * 16 + lo][h * 32 + kbB + i];
            v8f cz = {};
            cz = __builtin_amdgcn_wmma_f32_16x16x32_f16(false, aq, false, bb, (short)0, cz, false, false);
            float4 xi = ximg[(c % CPV) * 32 + sub * 16 + lo];
#pragma unroll
            for (int i = 0; i < 8; ++i) {
                float t = gf[i].x * xi.x + gf[i].y * xi.y + gf[i].z * xi.z;
                float w = __expf(-gf[i].w * t * t);
                raw[sub][i] = cz[i] * invs * w;
            }
        }
        // online softmax update (rows split lower/upper half-wave per C layout)
#pragma unroll
        for (int i = 0; i < 8; ++i) {
            float rm = fmaxf(raw[0][i], raw[1][i]);
            for (int off = 1; off < 16; off <<= 1) rm = fmaxf(rm, __shfl_xor(rm, off, 16));
            float Mn = fmaxf(Mrun[i], rm);
            float alpha = __expf(Mrun[i] - Mn);
            float p0 = __expf(raw[0][i] - Mn);
            float p1 = __expf(raw[1][i] - Mn);
            float ps = p0 + p1;
            for (int off = 1; off < 16; off <<= 1) ps += __shfl_xor(ps, off, 16);
            Srun[i] = Srun[i] * alpha + ps;
            Mrun[i] = Mn;
            O0[i] *= alpha;
            O1[i] *= alpha;
            pbuf[h][i + hi8][lo]      = (_Float16)p0;
            pbuf[h][i + hi8][lo + 16] = (_Float16)p1;
        }
        // p tile as A fragment (intra-wave LDS transpose; HW waits on DScnt)
        v16h ap;
#pragma unroll
        for (int i = 0; i < 8; ++i) ap[i]     = pbuf[h][lo][kbA + i];
#pragma unroll
        for (int i = 0; i < 8; ++i) ap[8 + i] = pbuf[h][lo][kbA + 8 + i];
        // V fragments from shared tile
        v16h bv0, bv1;
#pragma unroll
        for (int i = 0; i < 16; ++i) {
            bv0[i] = vTile[buf][kbB + i][h * 32 + lo];
            bv1[i] = vTile[buf][kbB + i][h * 32 + 16 + lo];
        }
        O0 = __builtin_amdgcn_wmma_f32_16x16x32_f16(false, ap, false, bv0, (short)0, O0, false, false);
        O1 = __builtin_amdgcn_wmma_f32_16x16x32_f16(false, ap, false, bv1, (short)0, O1, false, false);
        __syncthreads();
    }

#pragma unroll
    for (int i = 0; i < 8; ++i) {
        int qr = q0 + i + hi8;
        if (qr >= QQ) continue;
        float inv = 1.f / Srun[i];
        _Float16* op = attn_outH + ((size_t)(b * QQ + qr)) * 128 + h * 32;
        op[lo]      = (_Float16)(O0[i] * inv);
        op[16 + lo] = (_Float16)(O1[i] * inv);
    }
}

// ---------------------------------------------------------------------------
// launch
// ---------------------------------------------------------------------------
extern "C" void kernel_launch(void* const* d_in, const int* in_sizes, int n_in,
                              void* d_out, int out_size, void* d_ws, size_t ws_size,
                              hipStream_t stream) {
    (void)in_sizes; (void)n_in; (void)out_size; (void)ws_size;
    const float* x        = (const float*)d_in[0];
    const float* feature  = (const float*)d_in[1];
    const float* I_inv    = (const float*)d_in[2];
    const float* E_inv    = (const float*)d_in[3];
    const float* bev      = (const float*)d_in[4];
    const float* iplane   = (const float*)d_in[5];
    const float* bn_v_g   = (const float*)d_in[6];
    const float* bn_v_b   = (const float*)d_in[7];
    const float* bn_v_m   = (const float*)d_in[8];
    const float* bn_v_var = (const float*)d_in[9];
    const float* conv_v_w = (const float*)d_in[10];
    const float* bn_k_g   = (const float*)d_in[11];
    const float* bn_k_b   = (const float*)d_in[12];
    const float* bn_k_m   = (const float*)d_in[13];
    const float* bn_k_var = (const float*)d_in[14];
    const float* conv_k_w = (const float*)d_in[15];
    const float* ln_q_g   = (const float*)d_in[16];
    const float* ln_q_b   = (const float*)d_in[17];
    const float* ln_k_g   = (const float*)d_in[18];
    const float* ln_k_b   = (const float*)d_in[19];
    const float* ln_v_g   = (const float*)d_in[20];
    const float* ln_v_b   = (const float*)d_in[21];
    const float* Wq = (const float*)d_in[22]; const float* bq = (const float*)d_in[23];
    const float* Wk = (const float*)d_in[24]; const float* bk = (const float*)d_in[25];
    const float* Wv = (const float*)d_in[26]; const float* bv = (const float*)d_in[27];
    const float* Wo = (const float*)d_in[28]; const float* bo = (const float*)d_in[29];
    const float* ln_pre_g = (const float*)d_in[30];
    const float* ln_pre_b = (const float*)d_in[31];
    const float* W1 = (const float*)d_in[32]; const float* b1 = (const float*)d_in[33];
    const float* W2 = (const float*)d_in[34]; const float* b2 = (const float*)d_in[35];
    const float* ln_post_g = (const float*)d_in[36];
    const float* ln_post_b = (const float*)d_in[37];
    float* out = (float*)d_out;

    char* w = (char*)d_ws;
    size_t off = 0;
    auto alloc = [&](size_t bytes) { size_t o = off; off += (bytes + 255) & ~(size_t)255; return o; };

    _Float16* wH_cv = (_Float16*)(w + alloc(DDM * CCH * 2));
    _Float16* wH_ck = (_Float16*)(w + alloc(DDM * CCH * 2));
    _Float16* wH_q  = (_Float16*)(w + alloc(DDM * DDM * 2));
    _Float16* wH_k  = (_Float16*)(w + alloc(DDM * DDM * 2));
    _Float16* wH_v  = (_Float16*)(w + alloc(DDM * DDM * 2));
    _Float16* wH_o  = (_Float16*)(w + alloc(DDM * DDM * 2));
    _Float16* wH_1  = (_Float16*)(w + alloc(2 * DDM * DDM * 2));
    _Float16* wH_2  = (_Float16*)(w + alloc(2 * DDM * DDM * 2));
    _Float16* featH_v = (_Float16*)(w + alloc((size_t)MKV * CCH * 2));
    _Float16* featH_k = (_Float16*)(w + alloc((size_t)MKV * CCH * 2));
    float*    v_raw   = (float*)(w + alloc((size_t)MKV * DDM * 4));
    float*    k_raw   = (float*)(w + alloc((size_t)MKV * DDM * 4));
    _Float16* v_lnH   = (_Float16*)(w + alloc((size_t)MKV * DDM * 2));
    _Float16* k_lnH   = (_Float16*)(w + alloc((size_t)MKV * DDM * 2));
    _Float16* vpH     = (_Float16*)(w + alloc((size_t)MKV * DDM * 2));
    _Float16* kpH     = (_Float16*)(w + alloc((size_t)MKV * DDM * 2));
    _Float16* q_lnH   = (_Float16*)(w + alloc((size_t)MQ * DDM * 2));
    _Float16* qpH     = (_Float16*)(w + alloc((size_t)MQ * DDM * 2));
    float*    Eb      = (float*)(w + alloc(BNIMG * 12 * 4));
    float*    Ib      = (float*)(w + alloc(BNIMG * 9 * 4));
    float*    dmax    = (float*)(w + alloc(NBATCH * 4));
    float4*   G       = (float4*)(w + alloc((size_t)BNIMG * QQ * 16));
    float4*   xim     = (float4*)(w + alloc((size_t)KPP * 16));
    _Float16* attn_oH = (_Float16*)(w + alloc((size_t)MQ * DDM * 2));
    float*    o_raw   = (float*)(w + alloc((size_t)MQ * DDM * 4));
    float*    z_ln    = (float*)(w + alloc((size_t)MQ * DDM * 4));
    _Float16* z_lnH   = (_Float16*)(w + alloc((size_t)MQ * DDM * 2));
    _Float16* h1H     = (_Float16*)(w + alloc((size_t)MQ * 2 * DDM * 2));
    float*    m2      = (float*)(w + alloc((size_t)MQ * DDM * 4));

    auto conv16 = [&](const float* src, _Float16* dst, int n) {
        f32_to_f16_kernel<<<(n + 255) / 256, 256, 0, stream>>>(src, dst, n);
    };
    conv16(conv_v_w, wH_cv, DDM * CCH);
    conv16(conv_k_w, wH_ck, DDM * CCH);
    conv16(Wq, wH_q, DDM * DDM);
    conv16(Wk, wH_k, DDM * DDM);
    conv16(Wv, wH_v, DDM * DDM);
    conv16(Wo, wH_o, DDM * DDM);
    conv16(W1, wH_1, 2 * DDM * DDM);
    conv16(W2, wH_2, 2 * DDM * DDM);

    int btot = BNIMG * CCH * KPP;
    bn_relu_tr_kernel<<<(btot + 255) / 256, 256, 0, stream>>>(feature, bn_v_g, bn_v_b, bn_v_m, bn_v_var, featH_v);
    bn_relu_tr_kernel<<<(btot + 255) / 256, 256, 0, stream>>>(feature, bn_k_g, bn_k_b, bn_k_m, bn_k_var, featH_k);

    auto gemm = [&](const _Float16* X, const _Float16* Wm, const float* bias, void* Y,
                    int M, int Nn, int K, int act, int out_half) {
        int tiles = ((M + 15) / 16) * (Nn / 16);
        gemm_xwT_kernel<<<(tiles + 3) / 4, 128, 0, stream>>>(X, Wm, bias, Y, M, Nn, K, act, out_half);
    };

    gemm(featH_v, wH_cv, nullptr, v_raw, MKV, DDM, CCH, 0, 0);
    gemm(featH_k, wH_ck, nullptr, k_raw, MKV, DDM, CCH, 0, 0);

    ln128_kernel<<<(MKV + 3) / 4, 128, 0, stream>>>(v_raw, ln_v_g, ln_v_b, v_lnH, MKV);
    ln128_kernel<<<(MKV + 3) / 4, 128, 0, stream>>>(k_raw, ln_k_g, ln_k_b, k_lnH, MKV);
    ln_q_kernel<<<(MQ + 3) / 4, 128, 0, stream>>>(x, ln_q_g, ln_q_b, q_lnH);

    gemm(v_lnH, wH_v, bv, vpH, MKV, DDM, DDM, 0, 1);
    gemm(k_lnH, wH_k, bk, kpH, MKV, DDM, DDM, 0, 1);
    gemm(q_lnH, wH_q, bq, qpH, MQ, DDM, DDM, 0, 1);

    prep_mats_kernel<<<1, 32, 0, stream>>>(E_inv, I_inv, Eb, Ib);
    distmax_kernel<<<NBATCH, 256, 0, stream>>>(E_inv, bev, dmax);
    geom_kernel<<<(BNIMG * QQ + 255) / 256, 256, 0, stream>>>(bev, Eb, Ib, E_inv, dmax, G);
    ximg_kernel<<<(KPP + 255) / 256, 256, 0, stream>>>(iplane, xim);

    attn_kernel<<<dim3((QQ + 15) / 16, NBATCH), 128, 0, stream>>>(qpH, kpH, vpH, G, xim, attn_oH);

    gemm(attn_oH, wH_o, bo, o_raw, MQ, DDM, DDM, 0, 0);
    resid_ln_kernel<<<(MQ + 3) / 4, 128, 0, stream>>>(o_raw, x, ln_pre_g, ln_pre_b, z_ln, z_lnH);
    gemm(z_lnH, wH_1, b1, h1H, MQ, 2 * DDM, DDM, 2, 1);          // GELU
    gemm(h1H, wH_2, b2, m2, MQ, DDM, 2 * DDM, 0, 0);
    final_ln_kernel<<<(MQ + 3) / 4, 128, 0, stream>>>(z_ln, m2, ln_post_g, ln_post_b, out);
}
